// GatingNetwork_16638703305468
// MI455X (gfx1250) — compile-verified, matrix operand order checked by reading, not run on hardware
//
#include <hip/hip_runtime.h>
#include <hip/hip_bf16.h>
#include <math.h>
#include <stdint.h>

typedef unsigned int u32;
typedef unsigned short u16;
typedef unsigned long long u64;

typedef __bf16 bf16_t;
typedef bf16_t v16bf __attribute__((ext_vector_type(16)));
typedef float  v8f   __attribute__((ext_vector_type(8)));

#define K_DIM   65536
#define M_DIM   4096
#define N_OUT   200
#define N_PAD   224              // 14 tiles of 16
#define NT_WAVE 7                // N-tiles per wave (2 wave-groups cover 14)
#define KSTEP   32
#define NUM_KSTEPS (K_DIM / KSTEP)   // 2048
#define PF_DIST 8                // A-stream L2 prefetch distance (k-steps)

__device__ __forceinline__ u32 f2bf(float f) {
    u32 b = __float_as_uint(f);
    return (b + 0x7FFFu + ((b >> 16) & 1u)) >> 16;   // round-to-nearest-even
}
__device__ __forceinline__ u32 pack2(float lo, float hi) {
    return f2bf(lo) | (f2bf(hi) << 16);
}
// Flat->LDS byte address: LDS aperture keeps the LDS offset in addr[31:0]
__device__ __forceinline__ u32 lds_addr(const void* p) {
    return (u32)(uintptr_t)p;
}

union Frag {
    u32  u[8];
    v16bf v;
};

// ---------------------------------------------------------------------------
// w1 [K, 200] f32  ->  Bt [224, K] bf16  (transposed, zero-padded columns)
// ---------------------------------------------------------------------------
__global__ void k_convert_w1(const float* __restrict__ w1, u16* __restrict__ bt) {
    u32 idx = blockIdx.x * 256u + threadIdx.x;
    u32 k = idx & (K_DIM - 1);       // K_DIM == 2^16
    u32 n = idx >> 16;
    float v = (n < N_OUT) ? w1[(size_t)k * N_OUT + n] : 0.0f;
    bt[(size_t)n * K_DIM + k] = (u16)f2bf(v);
}

// ---------------------------------------------------------------------------
// h1[4096,200] = relu(x[4096,65536] @ w1 + b1), bf16 WMMA, fp32 accumulate
// block = 256 threads (8 waves). M=64 per block. grid = 64.
// wave -> (mStrip = wave&3) x (nSide = wave>>2, 7 N-tiles each)
// A tile (fp32->bf16 convert): global -> regs -> LDS
// B tile (already bf16):       global_load_async_to_lds_b128 (ASYNCcnt)
// ---------------------------------------------------------------------------
__global__ void __launch_bounds__(256) k_gemm1(
    const float* __restrict__ x, const u16* __restrict__ bt,
    const float* __restrict__ b1, float* __restrict__ h1)
{
    __shared__ __align__(16) u32 ldsA[2][64 * 16];     // 64 rows x 32 bf16
    __shared__ __align__(16) u32 ldsB[2][N_PAD * 16];  // 224 rows x 32 bf16

    const int tid    = threadIdx.x;
    const int lane   = tid & 31;
    const int wave   = tid >> 5;
    const int lrow   = lane & 15;
    const int hi     = lane >> 4;          // 0 or 1
    const int mStrip = wave & 3;
    const int nSide  = wave >> 2;          // 0: tiles 0..6, 1: tiles 7..13
    const int blockM = blockIdx.x * 64;

    v8f acc[NT_WAVE];
    #pragma unroll
    for (int t = 0; t < NT_WAVE; ++t) acc[t] = {};

    // A prefetch registers: 512 float4 chunks / 256 threads = 2 per thread
    float4 aReg[2];

    auto loadGlobalA = [&](int ks) {
        const int k0 = ks * KSTEP;
        #pragma unroll
        for (int i = 0; i < 2; ++i) {
            int c = tid + i * 256;
            int row = c >> 3, col4 = c & 7;   // 8 float4 per 32-float row
            const float* g = x + (size_t)(blockM + row) * K_DIM + k0 + col4 * 4;
            aReg[i] = *(const float4*)g;
            if (ks + PF_DIST < NUM_KSTEPS)                 // warm L2 ahead
                __builtin_prefetch(g + PF_DIST * KSTEP, 0, 1);
        }
    };
    auto storeLdsA = [&](int buf) {
        #pragma unroll
        for (int i = 0; i < 2; ++i) {
            int c = tid + i * 256;
            int row = c >> 3, col4 = c & 7;
            u32 lo  = pack2(aReg[i].x, aReg[i].y);
            u32 hi2 = pack2(aReg[i].z, aReg[i].w);
            *(uint2*)&ldsA[buf][row * 16 + col4 * 2] = make_uint2(lo, hi2);
        }
    };
    // B tile: 224 rows x 4 b128-chunks = 896 / 256 threads = 3.5 -> 4 guarded
    auto issueAsyncB = [&](int ks, int buf) {
        const int k0 = ks * KSTEP;
        #pragma unroll
        for (int i = 0; i < 4; ++i) {
            int c = tid + i * 256;
            if (c < N_PAD * 4) {
                int row = c >> 2, c4 = c & 3;   // 4 x (8 bf16) per 32-elem row
                u32 ldst = lds_addr(&ldsB[buf][row * 16 + c4 * 4]);
                u64 gsrc = (u64)(uintptr_t)(bt + (size_t)row * K_DIM + k0 + c4 * 8);
                asm volatile("global_load_async_to_lds_b128 %0, %1, off"
                             :: "v"(ldst), "v"(gsrc) : "memory");
            }
        }
    };

    // prologue: chunk 0 into buffer 0
    issueAsyncB(0, 0);
    loadGlobalA(0);
    storeLdsA(0);
    asm volatile("s_wait_asynccnt 0x0" ::: "memory");
    __syncthreads();

    // 16-bit A-operand striping: lane<16 -> K{0..7,16..23}; lane>=16 -> K{8..15,24..31}
    const int aRowBase = (mStrip * 16 + lrow) * 16;
    const int fOff = hi * 4;               // in u32 units

    for (int ks = 0; ks < NUM_KSTEPS; ++ks) {
        const int p = ks & 1;
        if (ks + 1 < NUM_KSTEPS) {
            issueAsyncB(ks + 1, p ^ 1);    // buf p^1 free since barrier of ks-1
            loadGlobalA(ks + 1);
        }

        Frag a;
        *(uint4*)&a.u[0] = *(const uint4*)&ldsA[p][aRowBase + fOff];
        *(uint4*)&a.u[4] = *(const uint4*)&ldsA[p][aRowBase + 8 + fOff];
        #pragma unroll
        for (int t = 0; t < NT_WAVE; ++t) {
            const int n0 = (nSide * NT_WAVE + t) * 16;
            const int bRowBase = (n0 + lrow) * 16;
            Frag b;
            *(uint4*)&b.u[0] = *(const uint4*)&ldsB[p][bRowBase + fOff];
            *(uint4*)&b.u[4] = *(const uint4*)&ldsB[p][bRowBase + 8 + fOff];
            acc[t] = __builtin_amdgcn_wmma_f32_16x16x32_bf16(
                false, a.v, false, b.v, (short)0, acc[t], false, false);
        }
        __syncthreads();                   // everyone done reading buffers p
        if (ks + 1 < NUM_KSTEPS) storeLdsA(p ^ 1);
        asm volatile("s_wait_asynccnt 0x0" ::: "memory"); // my B writes landed
        __syncthreads();                   // buffers p^1 published
    }

    // epilogue: bias + relu; C layout: VGPR r holds M=r (lanes 0-15) / M=r+8 (16-31)
    const int mBase = blockM + mStrip * 16 + hi * 8;
    #pragma unroll
    for (int t = 0; t < NT_WAVE; ++t) {
        const int n = (nSide * NT_WAVE + t) * 16 + lrow;
        if (n < N_OUT) {
            const float bias = b1[n];
            #pragma unroll
            for (int r = 0; r < 8; ++r) {
                float v = acc[t][r] + bias;
                h1[(size_t)(mBase + r) * N_OUT + n] = v > 0.0f ? v : 0.0f;
            }
        }
    }
}

// ---------------------------------------------------------------------------
// per-token: h2 = relu(h1@w2+b2); h3 = relu(h2@w3+b3)   (tiny, L2-resident)
// ---------------------------------------------------------------------------
__global__ void __launch_bounds__(256) k_mlp(
    const float* __restrict__ h1,
    const float* __restrict__ w2, const float* __restrict__ b2,
    const float* __restrict__ w3, const float* __restrict__ b3,
    float* __restrict__ h3)
{
    __shared__ float s1[200];
    __shared__ float s2[200];
    const int t = blockIdx.x;
    const int tid = threadIdx.x;
    if (tid < 200) s1[tid] = h1[(size_t)t * 200 + tid];
    __syncthreads();
    if (tid < 200) {
        float acc = b2[tid];
        for (int k = 0; k < 200; ++k) acc = fmaf(s1[k], w2[k * 200 + tid], acc);
        s2[tid] = acc > 0.f ? acc : 0.f;
    }
    __syncthreads();
    if (tid < 10) {
        float acc = b3[tid];
        for (int k = 0; k < 200; ++k) acc = fmaf(s2[k], w3[k * 10 + tid], acc);
        h3[(size_t)t * 10 + tid] = acc > 0.f ? acc : 0.f;
    }
}

// ---------------------------------------------------------------------------
// per-token routing: noisy logits -> top-8 -> sparse softmax
// out_probs [4096,64] then indices-as-float [4096,8]
// ---------------------------------------------------------------------------
__global__ void __launch_bounds__(64) k_route(
    const float* __restrict__ h3, const float* __restrict__ noise_eps,
    const float* __restrict__ wr, const float* __restrict__ br,
    const float* __restrict__ wn, const float* __restrict__ bn,
    float* __restrict__ out_probs, float* __restrict__ out_idx)
{
    __shared__ float sh[10];
    __shared__ float nl[64];
    __shared__ float wk[64];
    __shared__ float topv[8];
    __shared__ int   topi[8];
    __shared__ float s_sum, s_max;
    const int t = blockIdx.x;
    const int e = threadIdx.x;
    if (e < 10) sh[e] = h3[(size_t)t * 10 + e];
    __syncthreads();
    float lg = br[e], nz = bn[e];
    #pragma unroll
    for (int k = 0; k < 10; ++k) {
        lg = fmaf(sh[k], wr[k * 64 + e], lg);
        nz = fmaf(sh[k], wn[k * 64 + e], nz);
    }
    float sp = (nz > 20.f) ? nz : log1pf(expf(nz));   // softplus
    float noisy = lg + noise_eps[(size_t)t * 64 + e] * sp;
    nl[e] = noisy;
    wk[e] = noisy;
    __syncthreads();
    if (e == 0) {
        for (int j = 0; j < 8; ++j) {
            float best = -INFINITY; int bi = 0;
            for (int i = 0; i < 64; ++i)
                if (wk[i] > best) { best = wk[i]; bi = i; }
            topv[j] = best; topi[j] = bi; wk[bi] = -INFINITY;
        }
        float mx = topv[0];
        float sum = 0.f;
        for (int j = 0; j < 8; ++j) sum += expf(topv[j] - mx);
        s_sum = sum; s_max = mx;
    }
    __syncthreads();
    float p = 0.f;
    #pragma unroll
    for (int j = 0; j < 8; ++j)
        if (topi[j] == e) p = expf(nl[e] - s_max) / s_sum;
    out_probs[(size_t)t * 64 + e] = p;
    if (e < 8) out_idx[(size_t)t * 8 + e] = (float)topi[e];
}

// ---------------------------------------------------------------------------
extern "C" void kernel_launch(void* const* d_in, const int* in_sizes, int n_in,
                              void* d_out, int out_size, void* d_ws, size_t ws_size,
                              hipStream_t stream) {
    const float* x   = (const float*)d_in[0];
    const float* eps = (const float*)d_in[1];
    const float* w1  = (const float*)d_in[2];
    const float* b1  = (const float*)d_in[3];
    const float* w2  = (const float*)d_in[4];
    const float* b2  = (const float*)d_in[5];
    const float* w3  = (const float*)d_in[6];
    const float* b3  = (const float*)d_in[7];
    const float* wr  = (const float*)d_in[8];
    const float* br  = (const float*)d_in[9];
    const float* wn  = (const float*)d_in[10];
    const float* bn  = (const float*)d_in[11];

    char* ws = (char*)d_ws;
    const size_t btBytes = (size_t)N_PAD * K_DIM * sizeof(u16);   // 29,360,128
    const size_t h1Bytes = (size_t)M_DIM * N_OUT * sizeof(float); //  3,276,800
    u16*   bt = (u16*)ws;
    float* h1 = (float*)(ws + btBytes);
    float* h3 = (float*)(ws + btBytes + h1Bytes);

    float* outP = (float*)d_out;
    float* outI = outP + (size_t)M_DIM * 64;

    k_convert_w1<<<(N_PAD * K_DIM) / 256, 256, 0, stream>>>(w1, bt);
    k_gemm1<<<M_DIM / 64, 256, 0, stream>>>(x, bt, b1, h1);
    k_mlp<<<M_DIM, 256, 0, stream>>>(h1, w2, b2, w3, b3, h3);
    k_route<<<M_DIM, 64, 0, stream>>>(h3, eps, wr, br, wn, bn, outP, outI);
}
